// FactorizedTemporalVQVAE_71047349010703
// MI455X (gfx1250) — compile-verified, matrix operand order checked by reading, not run on hardware
//
#include <hip/hip_runtime.h>
#include <math.h>

// ---------------------------------------------------------------------------
// FactorizedTemporalVQVAE forward for gfx1250 (MI455X).
// GEMMs: v_wmma_f32_16x16x32_bf16, double-buffered LDS staging (one barrier
// per K-tile), 128-bit ds fragment loads, weights pre-transposed to bf16.
// B-tile staging uses GLOBAL_LOAD_ASYNC_TO_LDS_B128 (ASYNCcnt-tracked) when
// the toolchain exposes the builtin, else a synchronous b128 copy.
// ---------------------------------------------------------------------------

#define BROWS 8192
#define CHUNK_D 3344
#define E_DIM 128

typedef __attribute__((ext_vector_type(16))) __bf16 v16bf;
typedef __attribute__((ext_vector_type(8)))  __bf16 bf16x8;
typedef __attribute__((ext_vector_type(8)))  float  v8f;
typedef __attribute__((ext_vector_type(4)))  float  f32x4;
typedef __attribute__((ext_vector_type(4)))  int    i32x4;

union FragU { v16bf v; bf16x8 h[2]; };

#if defined(__gfx1250__) && \
    __has_builtin(__builtin_amdgcn_global_load_async_to_lds_b128) && \
    __has_builtin(__builtin_amdgcn_s_wait_asynccnt)
#define USE_ASYNC_B 1
#else
#define USE_ASYNC_B 0
#endif

#if USE_ASYNC_B
typedef __attribute__((address_space(1))) i32x4 g1_i32x4;
typedef __attribute__((address_space(3))) i32x4 l3_i32x4;
__device__ __forceinline__ void async_copy_b128(void* lds, const void* g) {
    __builtin_amdgcn_global_load_async_to_lds_b128(
        (g1_i32x4*)g, (l3_i32x4*)lds, 0, 0);
}
#endif

__device__ __forceinline__ void wait_stage_then_sync() {
#if USE_ASYNC_B
    __builtin_amdgcn_s_wait_asynccnt(0);
#endif
    __syncthreads();
}

__device__ __forceinline__ float gelu_exact(float x) {
    return 0.5f * x * (1.0f + erff(x * 0.70710678118654752440f));
}

// Epilogue modes
#define EP_NONE   0
#define EP_GELU   1
#define EP_RES    2   // out = acc + bias + res[row,col]
#define EP_SPLIT3 3   // decoder output: column n -> segment n/CHUNK of d_out

struct GemmArgs {
    const float*  A;     // [M,K] fp32 activations
    const __bf16* Wt;    // [N,K] bf16, pre-transposed weights
    const float*  bias;  // [N]
    const float*  res;   // [M,N] residual (EP_RES) or null
    float*        out;
    int M, N, K;
    int mode;
    int seg;             // CHUNK for EP_SPLIT3
};

// One-time weight prep: Wt[n][k] = (bf16) W[k][n].
__launch_bounds__(256)
__global__ void transpose_cvt_kernel(const float* __restrict__ W,
                                     __bf16* __restrict__ Wt, int K, int N) {
    __shared__ float tile[32][33];
    const int k0 = blockIdx.x * 32;
    const int n0 = blockIdx.y * 32;
    const int c  = threadIdx.x & 31;
    const int r8 = threadIdx.x >> 5;   // 0..7
#pragma unroll
    for (int i = 0; i < 4; ++i) {
        int k = k0 + r8 + i * 8;
        int n = n0 + c;
        tile[r8 + i * 8][c] = (k < K && n < N) ? W[(size_t)k * N + n] : 0.0f;
    }
    __syncthreads();
#pragma unroll
    for (int i = 0; i < 4; ++i) {
        int n = n0 + r8 + i * 8;
        int k = k0 + c;
        if (n < N && k < K) Wt[(size_t)n * K + k] = (__bf16)tile[c][r8 + i * 8];
    }
}

// ---- GEMM helpers --------------------------------------------------------

// Load 16 consecutive k of A (fp32) into regs, zero-padded past K.
__device__ __forceinline__ void load_a_regs(const float* aRow, int K, int k0,
                                            int ak, float fa[16]) {
    if (k0 + 64 <= K) {
        const float* ap = aRow + k0 + ak;
        *(f32x4*)&fa[0]  = *(const f32x4*)(ap + 0);
        *(f32x4*)&fa[4]  = *(const f32x4*)(ap + 4);
        *(f32x4*)&fa[8]  = *(const f32x4*)(ap + 8);
        *(f32x4*)&fa[12] = *(const f32x4*)(ap + 12);
    } else {
#pragma unroll
        for (int j = 0; j < 16; ++j) {
            int k = k0 + ak + j;
            fa[j] = (k < K) ? aRow[k] : 0.0f;
        }
    }
}

__device__ __forceinline__ void store_a_lds(__bf16 (*As)[72], int ar, int ak,
                                            const float fa[16]) {
    bf16x8 h0, h1;
#pragma unroll
    for (int j = 0; j < 4; ++j) {
        h0[j]     = (__bf16)fa[j];
        h0[j + 4] = (__bf16)fa[j + 4];
        h1[j]     = (__bf16)fa[j + 8];
        h1[j + 4] = (__bf16)fa[j + 12];
    }
    *(bf16x8*)&As[ar][ak]     = h0;
    *(bf16x8*)&As[ar][ak + 8] = h1;
}

// Stage 32 consecutive k of one B column (bf16) into LDS.
__device__ __forceinline__ void stage_b_lds(__bf16 (*Bs)[72],
                                            const __bf16* wRow, bool bOk,
                                            int K, int k0, int br, int bk) {
    if (bOk && (k0 + 64 <= K)) {
        const __bf16* wp = wRow + k0 + bk;
#if USE_ASYNC_B
#pragma unroll
        for (int j = 0; j < 4; ++j)
            async_copy_b128(&Bs[br][bk + 8 * j], wp + 8 * j);
#else
#pragma unroll
        for (int j = 0; j < 4; ++j)
            *(bf16x8*)&Bs[br][bk + 8 * j] = *(const bf16x8*)(wp + 8 * j);
#endif
    } else {
#pragma unroll
        for (int j = 0; j < 32; ++j) {
            int k = k0 + bk + j;
            Bs[br][bk + j] = (bOk && k < K) ? wRow[k] : (__bf16)0.0f;
        }
    }
}

__device__ __forceinline__ void compute_tile(const __bf16 (*As)[72],
                                             const __bf16 (*Bs)[72],
                                             int mw, int nw, int lane,
                                             v8f& acc00, v8f& acc01,
                                             v8f& acc10, v8f& acc11) {
#pragma unroll
    for (int s = 0; s < 2; ++s) {
        const int koff = s * 32 + ((lane >> 4) << 3);
        const int l15  = lane & 15;
        FragU a0, a1, b0, b1;
        a0.h[0] = *(const bf16x8*)&As[mw * 32 + l15][koff];
        a0.h[1] = *(const bf16x8*)&As[mw * 32 + l15][koff + 16];
        a1.h[0] = *(const bf16x8*)&As[mw * 32 + 16 + l15][koff];
        a1.h[1] = *(const bf16x8*)&As[mw * 32 + 16 + l15][koff + 16];
        b0.h[0] = *(const bf16x8*)&Bs[nw * 32 + l15][koff];
        b0.h[1] = *(const bf16x8*)&Bs[nw * 32 + l15][koff + 16];
        b1.h[0] = *(const bf16x8*)&Bs[nw * 32 + 16 + l15][koff];
        b1.h[1] = *(const bf16x8*)&Bs[nw * 32 + 16 + l15][koff + 16];
        acc00 = __builtin_amdgcn_wmma_f32_16x16x32_bf16(
            false, a0.v, false, b0.v, (short)0, acc00, false, false);
        acc01 = __builtin_amdgcn_wmma_f32_16x16x32_bf16(
            false, a0.v, false, b1.v, (short)0, acc01, false, false);
        acc10 = __builtin_amdgcn_wmma_f32_16x16x32_bf16(
            false, a1.v, false, b0.v, (short)0, acc10, false, false);
        acc11 = __builtin_amdgcn_wmma_f32_16x16x32_bf16(
            false, a1.v, false, b1.v, (short)0, acc11, false, false);
    }
}

// 256 threads = 8 wave32s. Macro-tile 64(M) x 128(N), K-step 64, LDS double
// buffered (one barrier per K-tile). Waves arranged 2x4: each wave owns a
// 32x32 slab = 4 accumulators, 8 WMMAs per staged tile.
__launch_bounds__(256)
__global__ void gemm_bf16_wmma_kernel(GemmArgs g) {
    // Row stride 72 elems = 144B: rows 16B-aligned, all fragment offsets are
    // multiples of 16B, and the 36-bank row step spreads a fragment's 16 rows
    // across distinct bank groups.
    __shared__ __align__(16) __bf16 As[2][64][72];
    __shared__ __align__(16) __bf16 Bs[2][128][72];

    const int tid  = threadIdx.x;
    const int lane = tid & 31;
    const int wave = tid >> 5;
    const int mw   = wave >> 2;    // 0..1
    const int nw   = wave & 3;     // 0..3
    const int bm   = blockIdx.y * 64;
    const int bn   = blockIdx.x * 128;

    // Fixed staging assignments (vector-width memory ops on the fast path).
    const int ar = tid >> 2;              // 0..63   (A row)
    const int ak = (tid & 3) << 4;        // 16 consecutive k
    const int br = tid >> 1;              // 0..127  (B column / Wt row)
    const int bk = (tid & 1) << 5;        // 32 consecutive k

    const float*  aRow = g.A  + (size_t)(bm + ar) * g.K;
    const __bf16* wRow = g.Wt + (size_t)(bn + br) * g.K;
    const bool    bOk  = (bn + br) < g.N;

    v8f acc00 = {}, acc01 = {}, acc10 = {}, acc11 = {};

    const int KT = (g.K + 63) >> 6;

    // Prologue: stage tile 0 into buffer 0.
    {
        float fa[16];
        load_a_regs(aRow, g.K, 0, ak, fa);
        store_a_lds(As[0], ar, ak, fa);
        stage_b_lds(Bs[0], wRow, bOk, g.K, 0, br, bk);
        wait_stage_then_sync();
    }

    for (int kt = 0; kt < KT; ++kt) {
        const int  cur       = kt & 1;
        const int  nxt       = cur ^ 1;
        const bool have_next = (kt + 1) < KT;
        const int  k0n       = (kt + 1) << 6;

        // 1) Issue next tile's global traffic first (A into regs; B straight
        //    into the other LDS buffer — async path bypasses VGPRs).
        float fa[16];
        if (have_next) {
            load_a_regs(aRow, g.K, k0n, ak, fa);
            stage_b_lds(Bs[nxt], wRow, bOk, g.K, k0n, br, bk);
        }
        // 2) Compute current tile while those loads are in flight.
        compute_tile(As[cur], Bs[cur], mw, nw, lane, acc00, acc01, acc10, acc11);
        // 3) Convert + store next A tile, then one barrier.
        if (have_next) store_a_lds(As[nxt], ar, ak, fa);
        wait_stage_then_sync();
    }

    // ---- epilogue (C/D layout: lanes 0-15 N=lane M=vgpr; 16-31 M=8+vgpr) ----
    const int l15   = lane & 15;
    const int rbase = bm + mw * 32 + ((lane >> 4) << 3);
#pragma unroll
    for (int tm = 0; tm < 2; ++tm) {
#pragma unroll
        for (int tn = 0; tn < 2; ++tn) {
            v8f acc = tm == 0 ? (tn == 0 ? acc00 : acc01)
                              : (tn == 0 ? acc10 : acc11);
            int col = bn + nw * 32 + tn * 16 + l15;
            if (col >= g.N) continue;
            int   row0 = rbase + tm * 16;
            float bia  = g.bias ? g.bias[col] : 0.0f;
#pragma unroll
            for (int r = 0; r < 8; ++r) {
                int   row = row0 + r;
                float v   = acc[r] + bia;
                if (g.mode == EP_GELU) v = gelu_exact(v);
                if (g.mode == EP_RES)  v += g.res[(size_t)row * g.N + col];
                if (g.mode == EP_SPLIT3) {
                    int seg = col / g.seg;
                    int c2  = col - seg * g.seg;
                    g.out[(size_t)seg * g.M * g.seg + (size_t)row * g.seg + c2] = v;
                } else {
                    g.out[(size_t)row * g.N + col] = v;
                }
            }
        }
    }
}

// LayerNorm (optionally fused GELU). One wave per row, 8 rows per block.
__launch_bounds__(256)
__global__ void ln_kernel(const float* __restrict__ x,
                          const float* __restrict__ gamma,
                          const float* __restrict__ beta,
                          float* __restrict__ out, int D, int do_gelu) {
    const int wave = threadIdx.x >> 5;
    const int lane = threadIdx.x & 31;
    const int row  = blockIdx.x * 8 + wave;
    const float* xr = x + (size_t)row * D;
    float s = 0.0f, ss = 0.0f;
    for (int c = lane; c < D; c += 32) {
        float v = xr[c];
        s += v; ss += v * v;
    }
#pragma unroll
    for (int o = 16; o > 0; o >>= 1) {
        s  += __shfl_xor(s,  o, 32);
        ss += __shfl_xor(ss, o, 32);
    }
    float mean = s / (float)D;
    float var  = ss / (float)D - mean * mean;   // ddof=0, matches jnp.var
    float inv  = rsqrtf(var + 1e-5f);
    float* orow = out + (size_t)row * D;
    for (int c = lane; c < D; c += 32) {
        float v = (xr[c] - mean) * inv * gamma[c] + beta[c];
        if (do_gelu) v = gelu_exact(v);
        orow[c] = v;
    }
}

// VQ: argmin_c ||z - cb[c]||^2 == argmin_c (cb.cb - 2 z.cb). One wave per row.
__launch_bounds__(256)
__global__ void vq_kernel(const float* __restrict__ z,
                          const float* __restrict__ cb, int ncodes,
                          float* __restrict__ zcat, int colofs,
                          float* __restrict__ idx_out) {
    const int wave = threadIdx.x >> 5;
    const int lane = threadIdx.x & 31;
    const int row  = blockIdx.x * 8 + wave;
    f32x4 zv = *(const f32x4*)(z + (size_t)row * E_DIM + lane * 4);
    float best = 3.4e38f;
    int   bi   = 0;
    for (int c = 0; c < ncodes; ++c) {
        f32x4 cv = *(const f32x4*)(cb + (size_t)c * E_DIM + lane * 4);
        float p = cv[0] * (cv[0] - 2.0f * zv[0]) + cv[1] * (cv[1] - 2.0f * zv[1])
                + cv[2] * (cv[2] - 2.0f * zv[2]) + cv[3] * (cv[3] - 2.0f * zv[3]);
#pragma unroll
        for (int o = 16; o > 0; o >>= 1) p += __shfl_xor(p, o, 32);
        if (p < best) { best = p; bi = c; }   // strict < keeps first min
    }
    f32x4 cq = *(const f32x4*)(cb + (size_t)bi * E_DIM + lane * 4);
    *(f32x4*)(zcat + (size_t)row * (3 * E_DIM) + colofs + lane * 4) = cq;
    if (lane == 0) idx_out[row] = (float)bi;
}

__global__ void set_scalar_kernel(float* p, float v) { *p = v; }

// ---------------------------------------------------------------------------
extern "C" void kernel_launch(void* const* d_in, const int* in_sizes, int n_in,
                              void* d_out, int out_size, void* d_ws, size_t ws_size,
                              hipStream_t stream) {
    (void)in_sizes; (void)n_in; (void)out_size; (void)ws_size;
    const float* pose     = (const float*)d_in[0];
    const float* motion   = (const float*)d_in[1];
    const float* dynamics = (const float*)d_in[2];
    // Params flattened as a JAX pytree: dict keys alphabetical, tuples (w,b)
    // in order. Top-level sorted: decoder, dyn_cb, dyn_enc, motion_cb,
    // motion_enc, pose_cb, pose_enc. Within an MLP: in(w,b), ln_b, ln_g,
    // out(w,b), rb_i{b, g, l1(w,b), l2(w,b)}.
    const float* const* P   = (const float* const*)(d_in + 3);
    const float* const* DEC = P + 0;    // 18 leaves
    const float* dyn_cb     = P[18];
    const float* const* DYN = P + 19;   // 12 leaves
    const float* motion_cb  = P[31];
    const float* const* MOT = P + 32;   // 12 leaves
    const float* pose_cb    = P[44];
    const float* const* POS = P + 45;   // 18 leaves

    float* out = (float*)d_out;
    const size_t BC = (size_t)BROWS * CHUNK_D;
    float* idx_pose   = out + 3 * BC;
    float* idx_motion = idx_pose + BROWS;
    float* idx_dyn    = idx_motion + BROWS;
    float* vq_loss    = idx_dyn + BROWS;

    // Workspace: activations first, then bf16 transposed weights.
    float* zcat = (float*)d_ws;                    // [B, 384]
    float* zbuf = zcat + (size_t)BROWS * 384;      // [B, 128]
    float* act0 = zbuf + (size_t)BROWS * 128;      // [B, 1024]
    float* act1 = act0 + (size_t)BROWS * 1024;     // [B, 1024]
    __bf16* wt_cursor = (__bf16*)(act1 + (size_t)BROWS * 1024);

    auto prep = [&](const float* W, int K, int N) -> const __bf16* {
        __bf16* dst = wt_cursor;
        wt_cursor += (size_t)K * N;
        transpose_cvt_kernel<<<dim3((K + 31) / 32, (N + 31) / 32),
                               dim3(256), 0, stream>>>(W, dst, K, N);
        return dst;
    };
    auto gemm = [&](const float* A, const float* W, const float* bias,
                    const float* res, float* o, int M, int N, int K, int mode) {
        GemmArgs g{A, prep(W, K, N), bias, res, o, M, N, K, mode, CHUNK_D};
        dim3 grid((N + 127) / 128, (M + 63) / 64);
        gemm_bf16_wmma_kernel<<<grid, dim3(256), 0, stream>>>(g);
    };
    auto ln = [&](const float* x, const float* gm, const float* bt, float* o,
                  int D, int do_gelu) {
        ln_kernel<<<dim3(BROWS / 8), dim3(256), 0, stream>>>(x, gm, bt, o, D,
                                                             do_gelu);
    };
    auto vq = [&](const float* z, const float* cb, int ncodes, int colofs,
                  float* idxo) {
        vq_kernel<<<dim3(BROWS / 8), dim3(256), 0, stream>>>(z, cb, ncodes,
                                                             zcat, colofs, idxo);
    };
    // Encoder: Q[0]=in.w Q[1]=in.b Q[2]=ln_b Q[3]=ln_g Q[4]=out.w Q[5]=out.b,
    // per-rb: rb[0]=b rb[1]=g rb[2]=l1.w rb[3]=l1.b rb[4]=l2.w rb[5]=l2.b
    auto encoder = [&](const float* x, const float* const* Q, int dH, int hH,
                       int n_rb, const float* cb, int ncodes, int colofs,
                       float* idxo) {
        gemm(x, Q[0], Q[1], nullptr, act0, BROWS, dH, CHUNK_D, EP_NONE);
        ln(act0, Q[3], Q[2], act0, dH, 1);  // gelu(ln(lin(x)))
        const float* const* rb = Q + 6;
        for (int i = 0; i < n_rb; ++i, rb += 6) {
            gemm(act0, rb[2], rb[3], nullptr, act1, BROWS, hH, dH, EP_GELU);
            gemm(act1, rb[4], rb[5], act0, act0, BROWS, dH, hH, EP_RES);
            ln(act0, rb[1], rb[0], act0, dH, 0);
        }
        gemm(act0, Q[4], Q[5], nullptr, zbuf, BROWS, E_DIM, dH, EP_NONE);
        vq(zbuf, cb, ncodes, colofs, idxo);
    };

    encoder(pose,     POS, 512, 1024, 2, pose_cb,   384, 0,   idx_pose);
    encoder(motion,   MOT, 512, 1024, 1, motion_cb, 192, 128, idx_motion);
    encoder(dynamics, DYN, 256, 512,  1, dyn_cb,    96,  256, idx_dyn);

    // Decoder
    gemm(zcat, DEC[0], DEC[1], nullptr, act0, BROWS, 512, 384, EP_NONE);
    ln(act0, DEC[3], DEC[2], act0, 512, 1);
    for (int i = 0; i < 2; ++i) {
        const float* const* rb = DEC + 6 + 6 * i;
        gemm(act0, rb[2], rb[3], nullptr, act1, BROWS, 1024, 512, EP_GELU);
        gemm(act1, rb[4], rb[5], act0, act0, BROWS, 512, 1024, EP_RES);
        ln(act0, rb[1], rb[0], act0, 512, 0);
    }
    // Output GEMM writes recon_pose/motion/dyn directly into d_out slices.
    gemm(act0, DEC[4], DEC[5], nullptr, out, BROWS, 3 * CHUNK_D, 512, EP_SPLIT3);

    set_scalar_kernel<<<1, 1, 0, stream>>>(vq_loss, 0.0f);
}
